// BiLSTM_CRF_16226386444288
// MI455X (gfx1250) — matrix operand order repaired and checked by reading, not compiled, not run on hardware
//
#include <hip/hip_runtime.h>
#include <math.h>

// Problem constants (from reference)
#define Vv 50000
#define Ee 256
#define Hh 256
#define Tt 20
#define Bb 64
#define Ll 512
#define START_ 18
#define STOP_ 19
#define NEGV -10000.0f

typedef __attribute__((ext_vector_type(16))) __bf16 v16bf;
typedef __attribute__((ext_vector_type(8)))  float  v8f;

__device__ __forceinline__ unsigned short f2bf(float f) {
  union { float f; unsigned u; } v; v.f = f;
  unsigned u = v.u;
  unsigned r = (u + 0x7FFFu + ((u >> 16) & 1u)) >> 16;  // round-nearest-even
  return (unsigned short)r;
}
__device__ __forceinline__ float sigm(float x) { return 1.0f / (1.0f + __expf(-x)); }

// ---------------------------------------------------------------------------
// Pack X (embedding gather, fwd + reversed) directly into WMMA A-fragment
// layout (ISA 7.12.2, 16-bit A 16x32):
//   element (m,k): hi=((k%16)>=8), lane=(m%16)+16*hi, slot=(k%8)+8*(k/16)
// xpack[d][t][mtile(4)][ktile(8)][lane(32)][slot(16)]  (bf16)
// ---------------------------------------------------------------------------
__global__ void pack_x_kernel(const int* __restrict__ sent,
                              const int* __restrict__ len,
                              const float* __restrict__ emb,
                              unsigned short* __restrict__ xpack) {
  unsigned id = blockIdx.x * 256u + threadIdx.x;   // < 2*512*64*256 = 2^24
  int e = id & 255;
  int b = (id >> 8) & 63;
  int t = (id >> 14) & 511;
  int d = id >> 23;
  int tt = t;
  if (d) {                                         // x_rev gather
    int r = len[b] - 1 - t;
    r = r < 0 ? 0 : (r > Ll - 1 ? Ll - 1 : r);
    tt = r;
  }
  int tok = sent[b * Ll + tt];
  float val = emb[(size_t)tok * Ee + e];
  int mt = b >> 4;
  int kt = e >> 5;
  int kk = e & 31;
  int hi = ((kk & 15) >= 8) ? 1 : 0;
  int lane = (b & 15) + (hi << 4);
  int s = (kk & 7) + ((kk >> 4) << 3);
  size_t dst = ((size_t)((d * Ll + t) * 4 + mt) * 8 + kt) * 512 + lane * 16 + s;
  xpack[dst] = f2bf(val);
}

// ---------------------------------------------------------------------------
// Pack W = [W_ih ; W_hh] (K=512, N=1024 gate units) into WMMA B-fragment
// layout (32x16 bf16): element (k,n): lane=(n%16)+16*(((k%32)%16)>=8), slot=(k%8)+8*((k%32)/16)
// wpack[d][ntile(64)][ktile(16)][lane(32)][slot(16)]
// ---------------------------------------------------------------------------
__global__ void pack_w_kernel(const float* __restrict__ wihf, const float* __restrict__ whhf,
                              const float* __restrict__ wihb, const float* __restrict__ whhb,
                              unsigned short* __restrict__ wpack) {
  unsigned id = blockIdx.x * 256u + threadIdx.x;   // < 2*1024*512 = 2^20
  int k = id & 511;
  int n = (id >> 9) & 1023;
  int d = id >> 19;
  const float* wih = d ? wihb : wihf;
  const float* whh = d ? whhb : whhf;
  float val = (k < Ee) ? wih[n * Ee + k] : whh[n * Hh + (k - Ee)];
  int nt = n >> 4;
  int kt = k >> 5;
  int kin = k & 31;
  int lane = (n & 15) + (((kin & 15) >= 8) ? 16 : 0);
  int s = (kin & 7) + ((kin >> 4) << 3);
  size_t dst = ((size_t)(d * 64 + nt) * 16 + kt) * 512 + lane * 16 + s;
  wpack[dst] = f2bf(val);
}

__global__ void init_state_kernel(float* __restrict__ h, float* __restrict__ c,
                                  unsigned short* __restrict__ hpack) {
  int id = blockIdx.x * 256 + threadIdx.x;         // 65536 threads
  hpack[id] = 0;                                   // both ping-pong parities
  if (id < 2 * Bb * Hh) { h[id] = 0.0f; c[id] = 0.0f; }
}

// ---------------------------------------------------------------------------
// One fused LSTM step for both directions.
// grid (8 wg, 2 dir), block 256 (8 waves).  Workgroup wg owns hidden units
// [32*wg, 32*wg+32): gate columns {u, 256+u, 512+u, 768+u} -> 8 N-tiles.
// GEMM: A = [x_t | h_{t-1}] (M=64, K=512, bf16 fragments), B = packed weights.
// Gates -> LDS -> pointwise update; new h written back in A-fragment layout
// (ping-pong on t parity to avoid cross-WG WAR hazard).
// ---------------------------------------------------------------------------
__global__ void __launch_bounds__(256) lstm_step_kernel(
    int t,
    const unsigned short* __restrict__ xpack,
    const unsigned short* __restrict__ wpack,
    unsigned short* __restrict__ hpack,            // 2 parities x 2 dir x 4 mt x 8 kt x 512
    float* __restrict__ h, float* __restrict__ c,
    float* __restrict__ hseq,
    const float* __restrict__ bihf, const float* __restrict__ bhhf,
    const float* __restrict__ bihb, const float* __restrict__ bhhb,
    const int* __restrict__ len) {
  __shared__ float gl[64 * 128];                   // 32 KB gate staging
  const int wg = blockIdx.x;                       // 0..7
  const int d  = blockIdx.y;                       // 0..1
  const int wave = threadIdx.x >> 5;
  const int lane = threadIdx.x & 31;
  const int rp = t & 1;                            // read parity
  const size_t PSTRIDE = (size_t)2 * 4 * 8 * 512;  // one parity (both dirs)

  const unsigned short* xbase = xpack + ((size_t)(d * Ll + t) * 4) * 8 * 512;
  const unsigned short* hrd   = hpack + (size_t)rp * PSTRIDE + (size_t)d * 4 * 8 * 512;
  unsigned short* hwr         = hpack + (size_t)(1 - rp) * PSTRIDE + (size_t)d * 4 * 8 * 512;
  const unsigned short* wbase = wpack + (size_t)d * 64 * 16 * 512;

  for (int j = 0; j < 4; ++j) {
    int ct = wave * 4 + j;
    int mt = ct & 3;                               // batch tile
    int ntl = ct >> 2;                             // 0..7 local N tile
    int group = ntl >> 1;                          // gate 0..3
    int sub = ntl & 1;
    int ntg = group * 16 + wg * 2 + sub;           // global N tile (of 64)
    v8f acc = {};
    #pragma unroll
    for (int kt = 0; kt < 16; ++kt) {
      const unsigned short* ap = (kt < 8)
          ? xbase + ((size_t)mt * 8 + kt) * 512 + lane * 16
          : hrd   + ((size_t)mt * 8 + (kt - 8)) * 512 + lane * 16;
      const unsigned short* bp = wbase + ((size_t)ntg * 16 + kt) * 512 + lane * 16;
      v16bf a  = *reinterpret_cast<const v16bf*>(ap);
      v16bf bm = *reinterpret_cast<const v16bf*>(bp);
      acc = __builtin_amdgcn_wmma_f32_16x16x32_bf16(false, a, false, bm,
                                                    (short)0, acc, false, false);
    }
    // C/D layout: VGPR r, lane l -> M = r + 8*(l/16), N = l%16
    int lr0 = mt * 16 + ((lane >> 4) << 3);
    int lc = ntl * 16 + (lane & 15);
    #pragma unroll
    for (int r = 0; r < 8; ++r) gl[(lr0 + r) * 128 + lc] = acc[r];
  }
  __syncthreads();

  const float* bih = d ? bihb : bihf;
  const float* bhh = d ? bhhb : bhhf;
  for (int q = 0; q < 8; ++q) {
    int idx = q * 256 + threadIdx.x;               // 0..2047 = 64 batch x 32 units
    int ul = idx & 31;
    int bb = idx >> 5;
    int u = wg * 32 + ul;
    float gi = gl[bb * 128 + ul]      + bih[u]       + bhh[u];
    float gf = gl[bb * 128 + 32 + ul] + bih[256 + u] + bhh[256 + u];
    float gg = gl[bb * 128 + 64 + ul] + bih[512 + u] + bhh[512 + u];
    float go = gl[bb * 128 + 96 + ul] + bih[768 + u] + bhh[768 + u];
    size_t sidx = (size_t)(d * Bb + bb) * Hh + u;
    float cOld = c[sidx];
    float hOld = h[sidx];
    float cn = sigm(gf) * cOld + sigm(gi) * tanhf(gg);
    float hn = sigm(go) * tanhf(cn);
    bool m = t < len[bb];
    float hK = m ? hn : hOld;
    float cK = m ? cn : cOld;
    h[sidx] = hK;
    c[sidx] = cK;
    hseq[((size_t)(d * Ll + t) * Bb + bb) * Hh + u] = m ? hn : 0.0f;
    // write new h into A-fragment layout for next step (k = 256+u -> ktile = 8+wg)
    int kk = ul;
    int hi2 = ((kk & 15) >= 8) ? 1 : 0;
    int lp = (bb & 15) + (hi2 << 4);
    int sp = (kk & 7) + ((kk >> 4) << 3);
    int mtp = bb >> 4;
    hwr[((size_t)mtp * 8 + wg) * 512 + lp * 16 + sp] = f2bf(hK);
  }
}

// ---------------------------------------------------------------------------
// Emissions: scores[b,t,tag] = b_out[tag] + [out_f(t) | out_b(gathered)]·w_out[tag]
// One wave per (b,t); stage the 512 concat features in LDS.
// ---------------------------------------------------------------------------
__global__ void emissions_kernel(const float* __restrict__ hseq,
                                 const int* __restrict__ len,
                                 const float* __restrict__ wout,
                                 const float* __restrict__ bout,
                                 float* __restrict__ scores) {
  __shared__ float hb[512];
  int pos = blockIdx.x;                            // b*512 + t
  int b = pos >> 9;
  int t = pos & 511;
  int lenb = len[b];
  int rev = lenb - 1 - t;
  rev = rev < 0 ? 0 : (rev > Ll - 1 ? Ll - 1 : rev);
  float mm = (t < lenb) ? 1.0f : 0.0f;
  const float* hf  = hseq + ((size_t)t * Bb + b) * Hh;
  const float* hbk = hseq + ((size_t)(Ll + rev) * Bb + b) * Hh;
  for (int u = threadIdx.x; u < Hh; u += 32) {
    hb[u] = hf[u];                                 // out_f already mask-zeroed
    hb[Hh + u] = hbk[u] * mm;
  }
  __syncthreads();
  if (threadIdx.x < Tt) {
    int tag = threadIdx.x;
    float s = bout[tag];
    for (int k = 0; k < 2 * Hh; ++k) s += hb[k] * wout[tag * (2 * Hh) + k];
    scores[(size_t)pos * Tt + tag] = s;
  }
}

// ---------------------------------------------------------------------------
// Viterbi: one wave per batch element. trans[i][j] = score(j -> i).
// ---------------------------------------------------------------------------
__global__ void viterbi_kernel(const float* __restrict__ scores,
                               const int* __restrict__ len,
                               const float* __restrict__ trans,
                               unsigned char* __restrict__ bps,
                               float* __restrict__ out) {
  __shared__ float fv[Tt];
  __shared__ float fvn[Tt];
  __shared__ float trS[Tt * Tt];
  __shared__ int bestIdx;
  int b = blockIdx.x;
  int lane = threadIdx.x;
  int lenb = len[b];
  for (int i = lane; i < Tt * Tt; i += 32) trS[i] = trans[i];
  if (lane < Tt) fv[lane] = (lane == START_) ? 0.0f : NEGV;
  __syncthreads();

  for (int t = 0; t < Ll; ++t) {
    float nf = 0.0f;
    if (lane < Tt) {
      float best = -3.4e38f;
      int bi = 0;
      for (int j = 0; j < Tt; ++j) {
        float v = fv[j] + trS[lane * Tt + j];
        if (v > best) { best = v; bi = j; }        // first-max (jnp.argmax)
      }
      bool m = t < lenb;
      float sc = scores[((size_t)b * Ll + t) * Tt + lane];
      nf = m ? (best + sc) : fv[lane];
      bps[((size_t)b * Ll + t) * Tt + lane] = (unsigned char)(m ? bi : STOP_);
    }
    __syncthreads();
    if (lane < Tt) fv[lane] = nf;
    __syncthreads();
  }

  if (lane < Tt) fvn[lane] = fv[lane] + trS[STOP_ * Tt + lane];
  __syncthreads();
  if (lane == 0) {
    float best = -3.4e38f;
    int bi = 0;
    for (int j = 0; j < Tt; ++j)
      if (fvn[j] > best) { best = fvn[j]; bi = j; }
    out[b] = best;
    bestIdx = bi;
  }
  __syncthreads();
  float* pout = out + Bb + (size_t)b * Ll;
  for (int p = lane; p < Ll; p += 32) pout[p] = 0.0f;
  __syncthreads();
  if (lane == 0) {
    int cur = bestIdx;
    pout[lenb - 1] = (float)cur;
    for (int j = lenb - 1; j >= 1; --j) {
      cur = bps[((size_t)b * Ll + j) * Tt + cur];
      pout[j - 1] = (float)cur;
    }
  }
}

// ---------------------------------------------------------------------------
extern "C" void kernel_launch(void* const* d_in, const int* in_sizes, int n_in,
                              void* d_out, int out_size, void* d_ws, size_t ws_size,
                              hipStream_t stream) {
  const int*   sent  = (const int*)d_in[0];
  const int*   len   = (const int*)d_in[1];
  const float* emb   = (const float*)d_in[2];
  const float* wihf  = (const float*)d_in[3];
  const float* whhf  = (const float*)d_in[4];
  const float* bihf  = (const float*)d_in[5];
  const float* bhhf  = (const float*)d_in[6];
  const float* wihb  = (const float*)d_in[7];
  const float* whhb  = (const float*)d_in[8];
  const float* bihb  = (const float*)d_in[9];
  const float* bhhb  = (const float*)d_in[10];
  const float* wout  = (const float*)d_in[11];
  const float* bout  = (const float*)d_in[12];
  const float* trans = (const float*)d_in[13];

  char* ws = (char*)d_ws;
  size_t off = 0;
  unsigned short* xpack = (unsigned short*)(ws + off); off += (size_t)2 * Ll * 4 * 8 * 512 * 2;  // 33.5 MB
  unsigned short* wpack = (unsigned short*)(ws + off); off += (size_t)2 * 64 * 16 * 512 * 2;     //  2 MB
  unsigned short* hpack = (unsigned short*)(ws + off); off += (size_t)2 * 2 * 4 * 8 * 512 * 2;   // 128 KB (2 parities)
  float* h      = (float*)(ws + off); off += (size_t)2 * Bb * Hh * 4;
  float* c      = (float*)(ws + off); off += (size_t)2 * Bb * Hh * 4;
  float* hseq   = (float*)(ws + off); off += (size_t)2 * Ll * Bb * Hh * 4;                       // 67 MB
  float* scores = (float*)(ws + off); off += (size_t)Bb * Ll * Tt * 4;
  unsigned char* bps = (unsigned char*)(ws + off); off += (size_t)Bb * Ll * Tt;

  pack_x_kernel<<<65536, 256, 0, stream>>>(sent, len, emb, xpack);
  pack_w_kernel<<<4096, 256, 0, stream>>>(wihf, whhf, wihb, whhb, wpack);
  init_state_kernel<<<256, 256, 0, stream>>>(h, c, hpack);
  for (int t = 0; t < Ll; ++t) {
    lstm_step_kernel<<<dim3(8, 2), 256, 0, stream>>>(t, xpack, wpack, hpack, h, c, hseq,
                                                     bihf, bhhf, bihb, bhhb, len);
  }
  emissions_kernel<<<Bb * Ll, 32, 0, stream>>>(hseq, len, wout, bout, scores);
  viterbi_kernel<<<Bb, 32, 0, stream>>>(scores, len, trans, bps, (float*)d_out);
}
